// StochasticBinaryConv2d_10419590660584
// MI455X (gfx1250) — matrix operand order repaired and checked by reading, not compiled, not run on hardware
//
#include <hip/hip_runtime.h>
#include <hip/hip_bf16.h>
#include <stdint.h>

// ---------------- problem constants ----------------
#define BB 4
#define ICN 64
#define OCN 64
#define HH 32
#define WW 32
#define NELEM (BB*OCN*HH*WW)   // 262144 per output tensor
#define NHALF (NELEM/2)        // 131072 (threefry counter split)
#define LOG2E 1.44269504088896340736f

typedef _Float16 h8  __attribute__((ext_vector_type(8)));
typedef _Float16 v16h __attribute__((ext_vector_type(16)));
typedef float    v8f  __attribute__((ext_vector_type(8)));

// ---------------- workspace layout (bytes) ----------------
#define OFF_XTP   0          // f16 xTp[b][34][34][64], zero-padded border: 591872 B
#define OFF_WH    591872     // f16 Wh[tap(ky*3+kx)][oc][ic]: 73728 B
#define OFF_E     665600     // f32 E[tap(l1*3+l2)][oc][ic] = exp(+2*Kf): 147456 B
#define OFF_EINV  813056     // f32 Einv: 147456 B
#define OFF_U     960512     // f32 u = exp(-a0), NELEM floats: 1048576 B
// total ~2.0 MB

// ---------------- threefry2x32-20, key = (0, 42) (jax.random.key(42)) -------
__device__ __forceinline__ uint32_t rotl32(uint32_t v, int s) {
    return (v << s) | (v >> (32 - s));
}
__device__ __forceinline__ void tf_round(uint32_t& x0, uint32_t& x1, int r) {
    x0 += x1; x1 = rotl32(x1, r); x1 ^= x0;
}
__device__ __forceinline__ uint2 threefry2x32(uint32_t c0, uint32_t c1) {
    const uint32_t k0 = 0u, k1 = 42u;
    const uint32_t k2 = 0x1BD11BDAu ^ k0 ^ k1;
    uint32_t x0 = c0 + k0, x1 = c1 + k1;
    tf_round(x0,x1,13); tf_round(x0,x1,15); tf_round(x0,x1,26); tf_round(x0,x1, 6);
    x0 += k1; x1 += k2 + 1u;
    tf_round(x0,x1,17); tf_round(x0,x1,29); tf_round(x0,x1,16); tf_round(x0,x1,24);
    x0 += k2; x1 += k0 + 2u;
    tf_round(x0,x1,13); tf_round(x0,x1,15); tf_round(x0,x1,26); tf_round(x0,x1, 6);
    x0 += k0; x1 += k1 + 3u;
    tf_round(x0,x1,17); tf_round(x0,x1,29); tf_round(x0,x1,16); tf_round(x0,x1,24);
    x0 += k1; x1 += k2 + 4u;
    tf_round(x0,x1,13); tf_round(x0,x1,15); tf_round(x0,x1,26); tf_round(x0,x1, 6);
    x0 += k2; x1 += k0 + 5u;
    return make_uint2(x0, x1);
}

// ---------------- prep: pad/transpose x -> f16, weights -> f16, exp tables --
__global__ void __launch_bounds__(256) prep_kernel(
    const float* __restrict__ x, const float* __restrict__ weight,
    _Float16* __restrict__ xTp, _Float16* __restrict__ Wh,
    float* __restrict__ E, float* __restrict__ Einv)
{
    const int N_XTP = BB*34*34*ICN;   // 295936
    const int N_W   = 9*OCN*ICN;      // 36864
    int t = blockIdx.x * 256 + threadIdx.x;
    if (t < N_XTP) {
        int ic = t & 63;
        int px = (t >> 6) % 34;
        int py = ((t >> 6) / 34) % 34;
        int b  = (t >> 6) / (34*34);
        float v = 0.0f;
        if (py >= 1 && py <= 32 && px >= 1 && px <= 32)
            v = x[((b*ICN + ic)*HH + (py-1))*WW + (px-1)];
        xTp[t] = (_Float16)v;
    } else if (t < N_XTP + N_W) {
        int s = t - N_XTP;
        int ic = s & 63, oc = (s >> 6) & 63, tap = s >> 12;
        Wh[s] = (_Float16)weight[((oc*ICN + ic)*3 + tap/3)*3 + (tap%3)];
    } else if (t < N_XTP + 2*N_W) {
        int s = t - N_XTP - N_W;
        int ic = s & 63, oc = (s >> 6) & 63, tap = s >> 12;
        int l1 = tap/3, l2 = tap%3;
        // Kf[oc,ic,l1,l2] = weight[oc,ic,2-l1,2-l2]
        float w = weight[((oc*ICN + ic)*3 + (2-l1))*3 + (2-l2)];
        E[s]    = __builtin_amdgcn_exp2f( 2.0f*w*LOG2E);
        Einv[s] = __builtin_amdgcn_exp2f(-2.0f*w*LOG2E);
    }
}

// ---------------- conv via WMMA + fused sigmoid/exp/threefry epilogue -------
// One wave computes D(16 pixels x 16 oc). A = x-patch (16x32 f16), B = weights
// (32x16 f16), accumulate over 9 taps x 2 k-chunks = 18 v_wmma_f32_16x16x32_f16.
__global__ void __launch_bounds__(256) conv_wmma_kernel(
    const _Float16* __restrict__ xTp, const _Float16* __restrict__ Wh,
    const float* __restrict__ bias,
    float* __restrict__ newx_out, float* __restrict__ u_out)
{
    int wid  = (blockIdx.x * 256 + threadIdx.x) >> 5;  // 0..1023
    int lane = threadIdx.x & 31;
    int m    = lane & 15;      // A: pixel row / B: oc column / D: oc column
    int half = lane >> 4;

    int oc0 = (wid & 3) << 4;
    int x0  = ((wid >> 2) & 1) << 4;
    int y   = (wid >> 3) & 31;
    int b   = wid >> 8;

    union U16 { v16h v; h8 h[2]; };
    v8f acc = {};

    #pragma unroll
    for (int dy = 0; dy < 3; ++dy) {
        #pragma unroll
        for (int dx = 0; dx < 3; ++dx) {
            // padded coords: row y+dy (0..33), col x0+m+dx (0..33)
            const _Float16* arow = xTp + ((b*34 + (y+dy))*34 + (x0 + m + dx))*64;
            const _Float16* brow = Wh  + (((dy*3 + dx)*OCN + (oc0 + m)))*64;
            #pragma unroll
            for (int kc = 0; kc < 64; kc += 32) {
                // A 16x32 f16: lane holds K = kc + half*8 + {0..7} and kc+16+half*8+{0..7}
                U16 A;
                A.h[0] = *(const h8*)(arow + kc + half*8);
                A.h[1] = *(const h8*)(arow + kc + 16 + half*8);
                // B 32x16 f16: lane holds K = kc + half*8 + {0..7} and kc+16+half*8+{0..7}
                U16 Bv;
                Bv.h[0] = *(const h8*)(brow + kc + half*8);
                Bv.h[1] = *(const h8*)(brow + kc + 16 + half*8);
                acc = __builtin_amdgcn_wmma_f32_16x16x32_f16(
                          false, A.v, false, Bv.v, (short)0, acc, false, false);
            }
        }
    }

    // Epilogue: D lane layout -> oc = oc0+m, pixel x = x0 + half*8 + g
    int oc   = oc0 + m;
    float bi = bias[oc];
    int idx0 = ((b*OCN + oc)*HH + y)*WW + (x0 + half*8);

    float nxs[8], uus[8];
    #pragma unroll
    for (int g = 0; g < 8; ++g) {
        float a0 = acc[g] + bi;
        float u  = __builtin_amdgcn_exp2f(-a0 * LOG2E);      // e^{-a0}
        float p0 = __builtin_amdgcn_rcpf(1.0f + u);          // sigmoid(a0)
        int idx  = idx0 + g;
        uint32_t c0 = (idx < NHALF) ? (uint32_t)idx : (uint32_t)(idx - NHALF);
        uint2 pr = threefry2x32(c0, c0 + (uint32_t)NHALF);
        uint32_t bits = (idx < NHALF) ? pr.x : pr.y;
        float uf = __uint_as_float((bits >> 9) | 0x3f800000u) - 1.0f;
        uf = fmaxf(uf, 0.0f);                                // jax uniform clamps at minval
        nxs[g] = (uf < p0) ? 1.0f : -1.0f;
        uus[g] = u;
    }
    *(float4*)(newx_out + idx0)     = make_float4(nxs[0], nxs[1], nxs[2], nxs[3]);
    *(float4*)(newx_out + idx0 + 4) = make_float4(nxs[4], nxs[5], nxs[6], nxs[7]);
    *(float4*)(u_out    + idx0)     = make_float4(uus[0], uus[1], uus[2], uus[3]);
    *(float4*)(u_out    + idx0 + 4) = make_float4(uus[4], uus[5], uus[6], uus[7]);
}

// ---------------- delta_q: per output, sum over 9 taps x 64 ic --------------
// sigma(a0 - 2*x*w) = 1/(1 + u * e^{2xw}); x=+-1 selects E or Einv.
// t = x*q staged in LDS (sign carries x, |t| = q, border = 0 kills invalid taps).
// Folding: sum (r - p0)*q = (sum r*q) - p0*(sum q).
__global__ void __launch_bounds__(256) deltaq_kernel(
    const float* __restrict__ x, const float* __restrict__ q,
    const float* __restrict__ E, const float* __restrict__ Einv,
    const float* __restrict__ u_in, const float* __restrict__ newx_in,
    float* __restrict__ newq_out)
{
    __shared__ float tq[64*102];   // [ic][row 0..2][col 0..33], 26112 B

    int ocg = blockIdx.x & 7;
    int Y   = (blockIdx.x >> 3) & 31;
    int b   = blockIdx.x >> 8;

    for (int i = threadIdx.x; i < 64*102; i += 256) {
        int ic  = i / 102;
        int rem = i - ic*102;
        int r   = rem / 34;
        int c   = rem - r*34;
        int gy  = Y - 1 + r;
        int gx  = c - 1;
        float v = 0.0f;
        if (gy >= 0 && gy < HH && gx >= 0 && gx < WW) {
            int gi = ((b*ICN + ic)*HH + gy)*WW + gx;
            v = x[gi] * q[gi];       // q >= 0, so sign(v) == sign(x), |v| == q
        }
        tq[i] = v;
    }
    __syncthreads();

    int X    = threadIdx.x & 31;          // wave-uniform oc, per-lane X
    int wave = threadIdx.x >> 5;
    int oc   = ocg*8 + wave;
    int idx  = ((b*OCN + oc)*HH + Y)*WW + X;

    float u  = u_in[idx];
    float p0 = __builtin_amdgcn_rcpf(1.0f + u);

    float acc = 0.0f, accq = 0.0f;
    #pragma unroll
    for (int l1 = 0; l1 < 3; ++l1) {
        #pragma unroll
        for (int l2 = 0; l2 < 3; ++l2) {
            const float* Er = E    + ((l1*3 + l2)*OCN + oc)*64;  // wave-uniform rows
            const float* Ir = Einv + ((l1*3 + l2)*OCN + oc)*64;
            int lb = (2 - l1)*34 + (X + 2 - l2);                 // LDS base for this tap
            #pragma unroll 4
            for (int ic = 0; ic < 64; ++ic) {
                float t  = tq[ic*102 + lb];
                float e  = (t >= 0.0f) ? Er[ic] : Ir[ic];
                float r  = __builtin_amdgcn_rcpf(fmaf(u, e, 1.0f)); // sigmoid(a_l)
                float aq = fabsf(t);
                acc  = fmaf(r, aq, acc);
                accq += aq;
            }
        }
    }
    // new_q = delta_q (delta_0 is exactly zero in forward value)
    float nx = newx_in[idx];
    newq_out[idx] = -nx * (acc - p0*accq);
}

// ---------------- launcher ----------------
extern "C" void kernel_launch(void* const* d_in, const int* in_sizes, int n_in,
                              void* d_out, int out_size, void* d_ws, size_t ws_size,
                              hipStream_t stream)
{
    const float* x      = (const float*)d_in[0];
    const float* q      = (const float*)d_in[1];
    const float* weight = (const float*)d_in[2];
    const float* bias   = (const float*)d_in[3];

    char* ws = (char*)d_ws;
    _Float16* xTp = (_Float16*)(ws + OFF_XTP);
    _Float16* Wh  = (_Float16*)(ws + OFF_WH);
    float*    E   = (float*)(ws + OFF_E);
    float*    Ei  = (float*)(ws + OFF_EINV);
    float*    U   = (float*)(ws + OFF_U);

    float* newx = (float*)d_out;
    float* newq = ((float*)d_out) + NELEM;

    const int prep_tasks = BB*34*34*ICN + 2*9*OCN*ICN;   // 369664
    prep_kernel<<<(prep_tasks + 255)/256, 256, 0, stream>>>(x, weight, xTp, Wh, E, Ei);
    conv_wmma_kernel<<<128, 256, 0, stream>>>(xTp, Wh, bias, newx, U);
    deltaq_kernel<<<1024, 256, 0, stream>>>(x, q, E, Ei, U, newx, newq);
}
